// Head_21663815041812
// MI455X (gfx1250) — matrix operand order repaired and checked by reading, not compiled, not run on hardware
//
#include <hip/hip_runtime.h>
#include <hip/hip_bf16.h>

typedef _Float16 f16_t;
typedef __attribute__((ext_vector_type(16))) _Float16 v16h;
typedef __attribute__((ext_vector_type(8)))  _Float16 v8h;
typedef __attribute__((ext_vector_type(8)))  float    v8f;
typedef __attribute__((ext_vector_type(4)))  int      v4i;

static constexpr int Bn = 4;
static constexpr int T  = 4096;
static constexpr int C  = 1024;
static constexpr int HS = 64;

#if defined(__has_builtin)
#if __has_builtin(__builtin_amdgcn_global_load_async_to_lds_b128) && \
    __has_builtin(__builtin_amdgcn_s_wait_asynccnt)
#define USE_ASYNC_LDS 1
#endif
#endif

__device__ __forceinline__ v8h ldv8(const f16_t* p) { return *(const v8h*)p; }

__device__ __forceinline__ v16h mk16(v8h lo, v8h hi) {
    v16h r;
#pragma unroll
    for (int e = 0; e < 8; ++e) { r[e] = lo[e]; r[e + 8] = hi[e]; }
    return r;
}

__device__ __forceinline__ v8f wmma_f16(v16h a, v16h b, v8f c) {
    // D = A(16x32 f16) * B(32x16 f16) + C(16x16 f32)
    return __builtin_amdgcn_wmma_f32_16x16x32_f16(
        /*neg_a=*/false, a, /*neg_b=*/false, b,
        /*c_mod=*/(short)0, c, /*reuse_a=*/false, /*reuse_b=*/false);
}

// Reduce across the 16 lanes holding one C-fragment row (masks <16 stay in half).
__device__ __forceinline__ float rowmax16(float v) {
#pragma unroll
    for (int m = 1; m < 16; m <<= 1) v = fmaxf(v, __shfl_xor(v, m, 32));
    return v;
}
__device__ __forceinline__ float rowsum16(float v) {
#pragma unroll
    for (int m = 1; m < 16; m <<= 1) v += __shfl_xor(v, m, 32);
    return v;
}

// A-fragment (16x32, f16) loaded from an fp32 row-major matrix, converted in regs.
// Lane mapping per CDNA5 ISA: lanes 0-15: K = e<8 ? e : e+8 ; lanes 16-31: +8.
__device__ __forceinline__ v16h load_a_from_f32(const float* row, int c, int half) {
    const int o0 = c + (half ? 8 : 0);
    float4 f0 = *(const float4*)(row + o0);
    float4 f1 = *(const float4*)(row + o0 + 4);
    float4 f2 = *(const float4*)(row + o0 + 16);
    float4 f3 = *(const float4*)(row + o0 + 20);
    v16h a;
    a[0]  = (f16_t)f0.x; a[1]  = (f16_t)f0.y; a[2]  = (f16_t)f0.z; a[3]  = (f16_t)f0.w;
    a[4]  = (f16_t)f1.x; a[5]  = (f16_t)f1.y; a[6]  = (f16_t)f1.z; a[7]  = (f16_t)f1.w;
    a[8]  = (f16_t)f2.x; a[9]  = (f16_t)f2.y; a[10] = (f16_t)f2.z; a[11] = (f16_t)f2.w;
    a[12] = (f16_t)f3.x; a[13] = (f16_t)f3.y; a[14] = (f16_t)f3.z; a[15] = (f16_t)f3.w;
    return a;
}

// ---------------- Kernel 0: weight fp32 -> f16 conversion ----------------
__global__ void cvt_w_kernel(const float* __restrict__ Wq, const float* __restrict__ Wk,
                             const float* __restrict__ Wv,
                             f16_t* __restrict__ Wqh, f16_t* __restrict__ Wkh,
                             f16_t* __restrict__ Wvh) {
    int i = blockIdx.x * blockDim.x + threadIdx.x;
    if (i < HS * C) {
        Wqh[i] = (f16_t)Wq[i];
        Wkh[i] = (f16_t)Wk[i];
        Wvh[i] = (f16_t)Wv[i];
    }
}

// ---------------- Kernel 1: fused QKV projection (WMMA) ----------------
// Block = 128 threads (4 waves). Each wave: 16 rows of x, all 64 output cols,
// for Q,K,V simultaneously (12 f32 accumulators). All 12 B-fragments are
// loaded into registers before the WMMA chain so loads clause-batch and
// overlap with matrix issue instead of wait(0)-serializing per fragment.
__global__ void __launch_bounds__(128)
qkv_proj_kernel(const float* __restrict__ x,
                const f16_t* __restrict__ Wqh, const f16_t* __restrict__ Wkh,
                const f16_t* __restrict__ Wvh,
                f16_t* __restrict__ Qh, f16_t* __restrict__ Kh, f16_t* __restrict__ Vh) {
    const int wave = threadIdx.x >> 5;
    const int lane = threadIdx.x & 31;
    const int lane16 = lane & 15;
    const int half = lane >> 4;
    const long tbase = (long)blockIdx.x * 64 + wave * 16;

    const f16_t* Ws[3] = {Wqh, Wkh, Wvh};
    v8f acc[12] = {};

    const float* xrow = x + (tbase + lane16) * (long)C;

    for (int c = 0; c < C; c += 32) {
        v16h a = load_a_from_f32(xrow, c, half);
        v16h bfrag[12];
#pragma unroll
        for (int m = 0; m < 3; ++m) {
#pragma unroll
            for (int j = 0; j < 4; ++j) {
                // B-fragment: lane -> col h = j*16+lane16; elems: K = e<8 ? e : e+8 ; lanes 16-31: +8
                const f16_t* wrow = Ws[m] + (j * 16 + lane16) * C + c + (half ? 8 : 0);
                bfrag[m * 4 + j] = mk16(ldv8(wrow), ldv8(wrow + 16));
            }
        }
#pragma unroll
        for (int i = 0; i < 12; ++i) acc[i] = wmma_f16(a, bfrag[i], acc[i]);
    }

    f16_t* Outs[3] = {Qh, Kh, Vh};
#pragma unroll
    for (int m = 0; m < 3; ++m)
#pragma unroll
        for (int j = 0; j < 4; ++j)
#pragma unroll
            for (int r = 0; r < 8; ++r) {
                long row = tbase + r + 8 * half;
                Outs[m][row * HS + j * 16 + lane16] = (f16_t)acc[m * 4 + j][r];
            }
}

// ---------------- Kernel 2: causal flash attention (WMMA) ----------------
// Block = 128 threads (4 waves), 64 queries per block (16 per wave).
// Key chunks of 32: K staged to LDS via GLOBAL_LOAD_ASYNC_TO_LDS_B128
// (ASYNCcnt path, no VGPR round-trip); V staged transposed through VGPRs.
__global__ void __launch_bounds__(128)
attn_kernel(const f16_t* __restrict__ Qh, const f16_t* __restrict__ Kh,
            const f16_t* __restrict__ Vh, float* __restrict__ out) {
    __shared__ __attribute__((aligned(16))) f16_t Ksh[32][HS];      // [key][h]
    __shared__ __attribute__((aligned(16))) f16_t Vts[HS][32];      // [h][key] (transposed)
    __shared__ __attribute__((aligned(16))) f16_t Plds[4][16][32];  // per-wave P round-trip

    const int tid = threadIdx.x;
    const int wave = tid >> 5;
    const int lane = tid & 31;
    const int lane16 = lane & 15;
    const int half = lane >> 4;

    const int blocksPerBatch = T / 64;
    const int b = blockIdx.x / blocksPerBatch;
    const int q0 = (blockIdx.x % blocksPerBatch) * 64;
    const long bT = (long)b * T;
    const int qbase = q0 + wave * 16;

    // Resident Q A-fragments (16 queries x 64 dims -> 2 K-steps of 32)
    v16h qa[2];
    {
        const f16_t* qrow = Qh + (bT + qbase + lane16) * HS;
#pragma unroll
        for (int ks = 0; ks < 2; ++ks) {
            int off = ks * 32 + (half ? 8 : 0);
            qa[ks] = mk16(ldv8(qrow + off), ldv8(qrow + off + 16));
        }
    }

    v8f o[4] = {};
    float mrow[8], lrow[8];
#pragma unroll
    for (int r = 0; r < 8; ++r) { mrow[r] = -__builtin_inff(); lrow[r] = 0.0f; }

    const int nkt = q0 / 32 + 2;  // chunks of 32 keys covering [0, q0+63]
    const float scale = 0.125f;   // 1/sqrt(64)

    for (int kt = 0; kt < nkt; ++kt) {
        const int n0 = kt * 32;

        // ---- cooperative staging ----
        {
#ifdef USE_ASYNC_LDS
            // K tile: 32 rows x 128B = 256 16-byte units; 2 async B128 per lane.
#pragma unroll
            for (int s = 0; s < 2; ++s) {
                const int u = tid + s * 128;
                const int krow = u >> 3;
                const int o16 = (u & 7) * 8;
                const f16_t* gsrc = Kh + (bT + n0 + krow) * HS + o16;
                __builtin_amdgcn_global_load_async_to_lds_b128(
                    (__attribute__((address_space(1))) v4i*)(v4i*)(void*)const_cast<f16_t*>(gsrc),
                    (__attribute__((address_space(3))) v4i*)(v4i*)(void*)&Ksh[krow][o16],
                    /*offset=*/0, /*cpol=*/0);
            }
#else
            const int krow = tid >> 2;
            const int kc0 = (tid & 3) * 16;
            const f16_t* ksrc = Kh + (bT + n0 + krow) * HS + kc0;
            *(v8h*)&Ksh[krow][kc0] = ldv8(ksrc);
            *(v8h*)&Ksh[krow][kc0 + 8] = ldv8(ksrc + 8);
#endif
            // V tile staged transposed (element reshuffle -> VGPR path).
            const int row = tid >> 2;
            const int c0 = (tid & 3) * 16;
            const f16_t* vsrc = Vh + (bT + n0 + row) * HS + c0;
            v8h v0 = ldv8(vsrc), v1 = ldv8(vsrc + 8);
#pragma unroll
            for (int e = 0; e < 8; ++e) {
                Vts[c0 + e][row] = v0[e];
                Vts[c0 + 8 + e][row] = v1[e];
            }
            if (kt + 1 < nkt) {  // global_prefetch_b8 of next chunk
                __builtin_prefetch(vsrc + 32 * HS, 0, 1);
            }
        }
#ifdef USE_ASYNC_LDS
        __builtin_amdgcn_s_wait_asynccnt(0);
#endif
        __syncthreads();

        if (n0 <= qbase + 15) {  // chunk not fully masked for this wave
            // ---- S = Q * K^T (two 16x16 N-tiles, two K-steps each) ----
            v8f s0 = {}, s1 = {};
#pragma unroll
            for (int ks = 0; ks < 2; ++ks) {
                const int hb = ks * 32 + (half ? 8 : 0);
                v16h b0 = mk16(ldv8(&Ksh[lane16][hb]), ldv8(&Ksh[lane16][hb + 16]));
                v16h b1 = mk16(ldv8(&Ksh[16 + lane16][hb]), ldv8(&Ksh[16 + lane16][hb + 16]));
                s0 = wmma_f16(qa[ks], b0, s0);
                s1 = wmma_f16(qa[ks], b1, s1);
            }

            // ---- scale, causal mask, online softmax ----
            const int key0 = n0 + lane16;
            const int key1 = key0 + 16;
#pragma unroll
            for (int r = 0; r < 8; ++r) {
                const int qg = qbase + r + 8 * half;
                float v0 = (key0 <= qg) ? s0[r] * scale : -__builtin_inff();
                float v1 = (key1 <= qg) ? s1[r] * scale : -__builtin_inff();
                float smax = rowmax16(fmaxf(v0, v1));
                float mnew = fmaxf(mrow[r], smax);
                float al = __expf(mrow[r] - mnew);
                float p0 = __expf(v0 - mnew);
                float p1 = __expf(v1 - mnew);
                lrow[r] = lrow[r] * al + rowsum16(p0 + p1);
                mrow[r] = mnew;
                o[0][r] *= al; o[1][r] *= al; o[2][r] *= al; o[3][r] *= al;
                // C-layout -> LDS (will be re-read in A-layout)
                const int rl = r + 8 * half;
                Plds[wave][rl][lane16] = (f16_t)p0;
                Plds[wave][rl][16 + lane16] = (f16_t)p1;
            }
            asm volatile("" ::: "memory");
#if __has_builtin(__builtin_amdgcn_s_wait_dscnt)
            __builtin_amdgcn_s_wait_dscnt(0);
#endif
            // ---- O += P * V (P as A-fragment, V^T rows feed B-fragments) ----
            const int ko = half ? 8 : 0;
            v16h pa = mk16(ldv8(&Plds[wave][lane16][ko]),
                           ldv8(&Plds[wave][lane16][16 + ko]));
            const int kb = half ? 8 : 0;
#pragma unroll
            for (int j = 0; j < 4; ++j) {
                v16h bv = mk16(ldv8(&Vts[j * 16 + lane16][kb]),
                               ldv8(&Vts[j * 16 + lane16][kb + 16]));
                o[j] = wmma_f16(pa, bv, o[j]);
            }
        }
        __syncthreads();
    }

    // ---- epilogue: O / l -> fp32 output ----
#pragma unroll
    for (int r = 0; r < 8; ++r) {
        const float inv = 1.0f / lrow[r];
        const long row = bT + qbase + r + 8 * half;
#pragma unroll
        for (int j = 0; j < 4; ++j)
            out[row * HS + j * 16 + lane16] = o[j][r] * inv;
    }
}

extern "C" void kernel_launch(void* const* d_in, const int* in_sizes, int n_in,
                              void* d_out, int out_size, void* d_ws, size_t ws_size,
                              hipStream_t stream) {
    // setup_inputs order: x, Wk, Wq, Wv, i, embed_dim, head_size_sel
    const float* x  = (const float*)d_in[0];
    const float* Wk = (const float*)d_in[1];
    const float* Wq = (const float*)d_in[2];
    const float* Wv = (const float*)d_in[3];
    // i / embed_dim / head_size_sel are fixed (0, 1024, 64) per the reference setup.

    char* ws = (char*)d_ws;
    f16_t* Wqh = (f16_t*)(ws + (size_t)0);
    f16_t* Wkh = (f16_t*)(ws + (size_t)(128 << 10));
    f16_t* Wvh = (f16_t*)(ws + (size_t)(256 << 10));
    f16_t* Qh  = (f16_t*)(ws + (size_t)(512 << 10));
    f16_t* Kh  = Qh + (size_t)Bn * T * HS;
    f16_t* Vh  = Kh + (size_t)Bn * T * HS;

    cvt_w_kernel<<<(HS * C + 255) / 256, 256, 0, stream>>>(Wq, Wk, Wv, Wqh, Wkh, Wvh);
    qkv_proj_kernel<<<(Bn * T) / 64, 128, 0, stream>>>(x, Wqh, Wkh, Wvh, Qh, Kh, Vh);
    attn_kernel<<<(Bn * T) / 64, 128, 0, stream>>>(Qh, Kh, Vh, (float*)d_out);
}
